// EdgeTransition_31112743092522
// MI455X (gfx1250) — matrix operand order repaired and checked by reading, not compiled, not run on hardware
//
#include <hip/hip_runtime.h>
#include <stdint.h>

#define N_NODES   20000
#define N_EDGES   640000
#define D_NODE    256
#define D_BIAS    128
#define D_EDGE    128
#define D_HID     384
#define D_OUT     128

#define EST       392          // padded LDS row stride (bf16) for 384-wide tiles
#define PST       392          // padded panel row stride (bf16), K=384
#define FST       132          // padded LDS row stride (f32) for 128-wide LN tile
#define MTILE     128          // rows (edges / nodes) per block
#define THREADS   256          // 8 waves (wave32)

// ---- shared memory layout (dynamic; ~295 KB, 1 workgroup / WGP) ----
#define SM_EBUF   0
#define SM_XBUF   (MTILE * EST * 2)                    // 100352
#define SM_PAN0   (2 * MTILE * EST * 2)                // 200704
#define SM_PAN1   (SM_PAN0 + 64 * PST * 2)             // 250880
#define SM_MU     (SM_PAN1 + 64 * PST * 2)             // 301056
#define SM_RS     (SM_MU + MTILE * 4)
#define SM_BYTES  (SM_RS + MTILE * 4)                  // 302080 <= 320K WGP LDS

typedef __attribute__((ext_vector_type(16))) __bf16       v16bf;
typedef __attribute__((ext_vector_type(8)))  float        v8f;
typedef __attribute__((ext_vector_type(4)))  unsigned int v4u;
typedef __attribute__((ext_vector_type(4)))  int          v4i;
typedef __attribute__((ext_vector_type(4)))  float        v4f;
typedef __attribute__((ext_vector_type(2)))  float        v2f;

union BF16x16 { v16bf v; v4u q[2]; unsigned short s[16]; };
union FU      { float f; unsigned int u; };

#if __has_builtin(__builtin_amdgcn_global_load_async_to_lds_b128)
#define HAS_ASYNC_LDS 1
#else
#define HAS_ASYNC_LDS 0
#endif

// async builtin pointer params are v4i* in explicit address spaces (per hipcc diag)
#define AS_GLOBAL_V4I(p) ((__attribute__((address_space(1))) v4i*)(p))
#define AS_LDS_V4I(p)    ((__attribute__((address_space(3))) v4i*)(p))

__device__ __forceinline__ void async_copy16(const void* gsrc, void* ldst) {
#if HAS_ASYNC_LDS
  __builtin_amdgcn_global_load_async_to_lds_b128(AS_GLOBAL_V4I(gsrc), AS_LDS_V4I(ldst), 0, 0);
#else
  *(v4u*)ldst = *(const v4u*)gsrc;
#endif
}

__device__ __forceinline__ void wait_async_all() {
#if __has_builtin(__builtin_amdgcn_s_wait_asynccnt)
  __builtin_amdgcn_s_wait_asynccnt(0);
#else
  asm volatile("s_wait_asynccnt 0x0" ::: "memory");
#endif
}

__device__ __forceinline__ unsigned short f2b(float f) {
  FU t; t.f = f;
  unsigned int u = t.u + 0x7FFFu + ((t.u >> 16) & 1u);   // round-to-nearest-even
  return (unsigned short)(u >> 16);
}
__device__ __forceinline__ float b2f(unsigned short s) {
  FU t; t.u = ((unsigned int)s) << 16; return t.f;
}
__device__ __forceinline__ unsigned int pack2(float lo, float hi) {
  return ((unsigned int)f2b(hi) << 16) | (unsigned int)f2b(lo);
}

// Load NK A-fragments (16x32 bf16 each) for this lane's row.
// A layout: lanes 0-15: M=lane, K=k0+{0..7,16..23}; lanes 16-31: M=lane-16, K=k0+{8..15,24..31}
template <int NK>
__device__ __forceinline__ void load_afrags(v16bf* af, const unsigned short* arow, int kb) {
#pragma unroll
  for (int kt = 0; kt < NK; ++kt) {
    BF16x16 t;
    t.q[0] = *(const v4u*)(arow + kt * 32 + kb);
    t.q[1] = *(const v4u*)(arow + kt * 32 + kb + 16);
    af[kt] = t.v;
  }
}

// One 16x16 tile: B-fragment loads pipelined against a WMMA accumulation chain
// (D->C chaining is the accumulate path; no D->A/B hazard NOPs needed).
template <int NK>
__device__ __forceinline__ v8f gemm_tile(const v16bf* af, const unsigned short* bcol, float bv) {
  v16bf bfr[NK];
#pragma unroll
  for (int kt = 0; kt < NK; ++kt) {
    BF16x16 b;
    b.q[0] = *(const v4u*)(bcol + kt * 32);
    b.q[1] = *(const v4u*)(bcol + kt * 32 + 16);
    bfr[kt] = b.v;
  }
  v8f acc;
#pragma unroll
  for (int i = 0; i < 8; ++i) acc[i] = bv;
#pragma unroll
  for (int kt = 0; kt < NK; ++kt)
    acc = __builtin_amdgcn_wmma_f32_16x16x32_bf16(false, af[kt], false, bfr[kt],
                                                  (short)0, acc, false, false);
  return acc;
}

// Stage a 64-col x K=384 bf16 weight panel (contiguous 49152B in global) into
// padded LDS (row stride 784B) with async global->LDS b128 (ASYNCcnt).
__device__ __forceinline__ void stage_panel(char* lds_base, const unsigned short* gsrc, int tid) {
#pragma unroll
  for (int i = 0; i < 12; ++i) {
    const int chunk = tid + i * THREADS;
    const int r = chunk / 48, c = chunk - r * 48;      // 48 x 16B per row
    async_copy16((const char*)gsrc + chunk * 16, lds_base + r * (PST * 2) + c * 16);
  }
}

// Pipelined panel GEMM: double-buffered async staging overlapped with compute.
// store(acc, n) writes one 16x16 f32 tile.
template <int NG, typename F>
__device__ __forceinline__ void gemm_panels(const v16bf* af, const unsigned short* wsrc,
                                            const float* __restrict__ bias,
                                            char* p0, char* p1,
                                            int tid, int lane16, int hi, F&& store) {
  stage_panel(p0, wsrc, tid);
  for (int g = 0; g < NG; ++g) {
    wait_async_all();          // this wave's async LDS writes for stage(g) done
    __syncthreads();           // all waves: stage(g) visible, compute(g-1) done
    char* cur = (g & 1) ? p1 : p0;
    if (g + 1 < NG)
      stage_panel((g & 1) ? p0 : p1, wsrc + (size_t)(g + 1) * 64 * D_HID, tid);
    const unsigned short* curHW = (const unsigned short*)cur;
    for (int nl = 0; nl < 4; ++nl) {
      const int n = (g * 4 + nl) * 16 + lane16;
      v8f acc = gemm_tile<12>(af, curHW + (nl * 16 + lane16) * PST + hi * 8, bias[n]);
      store(acc, n);
    }
  }
}

// ---------- prep: W[K][N] f32 -> Wt[N][K] bf16 ----------
__global__ void cvt_transpose_kernel(const float* __restrict__ src,
                                     unsigned short* __restrict__ dst, int K, int N) {
  const int idx = blockIdx.x * blockDim.x + threadIdx.x;
  if (idx >= K * N) return;
  const int n = idx / K, k = idx - n * K;
  dst[idx] = f2b(src[k * N + n]);
}

// ---------- h = node_embed @ W0 + b0 -> bf16 ----------
__global__ void __launch_bounds__(THREADS)
node_proj_kernel(const float* __restrict__ node_embed,
                 const unsigned short* __restrict__ w0t,
                 const float* __restrict__ b0,
                 unsigned short* __restrict__ h) {
  const int tid = threadIdx.x;
  const int wave = tid >> 5, lane = tid & 31;
  const int lane16 = lane & 15, hi = lane >> 4;
  const int m0 = blockIdx.x * MTILE + wave * 16;
  const int kb = hi * 8;
  const bool full = (m0 + 16) <= N_NODES;                // wave-uniform fast path

  int mrow = m0 + lane16;
  if (mrow > N_NODES - 1) mrow = N_NODES - 1;            // clamp loads; stores guarded
  const float* arow = node_embed + (size_t)mrow * D_NODE;

  v16bf af[8];                                           // K = 256 -> 8 steps of 32
#pragma unroll
  for (int kt = 0; kt < 8; ++kt) {
    const int k0 = kt * 32 + kb;
    v4f f0 = *(const v4f*)(arow + k0);
    v4f f1 = *(const v4f*)(arow + k0 + 4);
    v4f f2 = *(const v4f*)(arow + k0 + 16);
    v4f f3 = *(const v4f*)(arow + k0 + 20);
    BF16x16 t;
    t.s[0]  = f2b(f0.x); t.s[1]  = f2b(f0.y); t.s[2]  = f2b(f0.z); t.s[3]  = f2b(f0.w);
    t.s[4]  = f2b(f1.x); t.s[5]  = f2b(f1.y); t.s[6]  = f2b(f1.z); t.s[7]  = f2b(f1.w);
    t.s[8]  = f2b(f2.x); t.s[9]  = f2b(f2.y); t.s[10] = f2b(f2.z); t.s[11] = f2b(f2.w);
    t.s[12] = f2b(f3.x); t.s[13] = f2b(f3.y); t.s[14] = f2b(f3.z); t.s[15] = f2b(f3.w);
    af[kt] = t.v;
  }

  for (int nt = 0; nt < 8; ++nt) {
    const int n = nt * 16 + lane16;
    v8f acc = gemm_tile<8>(af, w0t + (size_t)n * D_NODE + hi * 8, b0[n]);
    if (full) {
#pragma unroll
      for (int r = 0; r < 8; ++r)
        h[(size_t)(m0 + r + hi * 8) * D_BIAS + n] = f2b(acc[r]);
    } else {
#pragma unroll
      for (int r = 0; r < 8; ++r) {
        const int m = m0 + r + hi * 8;
        if (m < N_NODES) h[(size_t)m * D_BIAS + n] = f2b(acc[r]);
      }
    }
  }
}

// ---------- fused edge kernel: gather -> MLP trunk -> residual -> final -> LN ----------
__global__ void __launch_bounds__(THREADS)
edge_trunk_kernel(const float* __restrict__ edge_embed,
                  const long long* __restrict__ eidx,
                  const unsigned short* __restrict__ h,
                  const unsigned short* __restrict__ w1t, const float* __restrict__ b1,
                  const unsigned short* __restrict__ w2t, const float* __restrict__ b2,
                  const unsigned short* __restrict__ wft, const float* __restrict__ bfi,
                  const float* __restrict__ gamma, const float* __restrict__ beta,
                  float* __restrict__ out) {
  extern __shared__ char smem[];
  unsigned short* eBuf = (unsigned short*)(smem + SM_EBUF);   // e, later r = relu(.)+e
  unsigned short* xBuf = (unsigned short*)(smem + SM_XBUF);   // x1
  float*          fBuf = (float*)(smem + SM_XBUF);            // overlays xBuf (dead in GEMM3)
  char*           pan0 = smem + SM_PAN0;
  char*           pan1 = smem + SM_PAN1;
  float*          muBuf = (float*)(smem + SM_MU);
  float*          rsBuf = (float*)(smem + SM_RS);

  const int tid = threadIdx.x;
  const long long e0 = (long long)blockIdx.x * MTILE;

  // Phase 0: e = [edge_embed | h[dst] | h[src]]  (bf16, padded rows)
  // h gathers are raw bf16 row copies -> async global->LDS b128 (16 chunks/row).
  // 16 consecutive lanes cover one 256B row: coalesced.
#pragma unroll
  for (int i = 0; i < 8; ++i) {                 // dst: cols 128..255
    const int idx = tid + i * THREADS;
    const int m = idx >> 4, c = idx & 15;
    const int nid = (int)eidx[e0 + m];
    async_copy16((const char*)(h + (size_t)nid * D_BIAS) + c * 16,
                 (char*)(eBuf + m * EST + 128) + c * 16);
  }
#pragma unroll
  for (int i = 0; i < 8; ++i) {                 // src: cols 256..383
    const int idx = tid + i * THREADS;
    const int m = idx >> 4, c = idx & 15;
    const int nid = (int)eidx[(long long)N_EDGES + e0 + m];
    async_copy16((const char*)(h + (size_t)nid * D_BIAS) + c * 16,
                 (char*)(eBuf + m * EST + 256) + c * 16);
  }
  // edge_embed: f32 -> bf16 conversion must go through VALU
  for (int idx = tid; idx < MTILE * 64; idx += THREADS) {
    const int m = idx >> 6, c2 = idx & 63;
    v2f f = *(const v2f*)(edge_embed + (size_t)(e0 + m) * D_EDGE + 2 * c2);
    ((unsigned int*)(eBuf + m * EST))[c2] = pack2(f.x, f.y);
  }
  wait_async_all();
  __syncthreads();

  const int wave = tid >> 5, lane = tid & 31;
  const int lane16 = lane & 15, hi = lane >> 4;
  const int wrow = wave * 16;
  const int kb = hi * 8;

  // ---- GEMM1: x1 = relu(e @ W1 + b1) ----
  {
    v16bf af[12];
    load_afrags<12>(af, eBuf + (wrow + lane16) * EST, kb);
    gemm_panels<6>(af, w1t, b1, pan0, pan1, tid, lane16, hi,
                   [&](const v8f& acc, int n) {
#pragma unroll
                     for (int r = 0; r < 8; ++r)
                       xBuf[(wrow + r + hi * 8) * EST + n] = f2b(fmaxf(acc[r], 0.f));
                   });
  }
  __syncthreads();

  // ---- GEMM2: r = relu(x1 @ W2 + b2) + e   (in place over eBuf; each (m,n) wave-owned) ----
  {
    v16bf af[12];
    load_afrags<12>(af, xBuf + (wrow + lane16) * EST, kb);
    gemm_panels<6>(af, w2t, b2, pan0, pan1, tid, lane16, hi,
                   [&](const v8f& acc, int n) {
#pragma unroll
                     for (int r = 0; r < 8; ++r) {
                       const int off = (wrow + r + hi * 8) * EST + n;
                       eBuf[off] = f2b(fmaxf(acc[r], 0.f) + b2f(eBuf[off]));
                     }
                   });
  }
  __syncthreads();

  // ---- GEMM3: o = r @ Wf + bf  (f32 tile for layernorm) ----
  {
    v16bf af[12];
    load_afrags<12>(af, eBuf + (wrow + lane16) * EST, kb);
    gemm_panels<2>(af, wft, bfi, pan0, pan1, tid, lane16, hi,
                   [&](const v8f& acc, int n) {
#pragma unroll
                     for (int r = 0; r < 8; ++r)
                       fBuf[(wrow + r + hi * 8) * FST + n] = acc[r];
                   });
  }
  __syncthreads();

  // ---- LayerNorm over last dim (biased var, eps=1e-5) ----
  if (tid < MTILE) {
    const float* row = fBuf + tid * FST;
    float s = 0.f, s2 = 0.f;
    for (int c = 0; c < D_OUT; ++c) { const float v = row[c]; s += v; s2 += v * v; }
    const float mu  = s * (1.f / D_OUT);
    const float var = s2 * (1.f / D_OUT) - mu * mu;
    muBuf[tid] = mu;
    rsBuf[tid] = rsqrtf(var + 1e-5f);
  }
  __syncthreads();
  for (int idx = tid; idx < MTILE * D_OUT; idx += THREADS) {
    const int m = idx >> 7, c = idx & 127;
    const float v = (fBuf[m * FST + c] - muBuf[m]) * rsBuf[m] * gamma[c] + beta[c];
    out[(size_t)(e0 + m) * D_OUT + c] = v;
  }
}

// ---------- workspace layout (bytes) ----------
#define WS_H    0                                  // 20000*128*2  = 5,120,000
#define WS_W0T  5120000                            // 128*256*2    =    65,536
#define WS_W1T  5185536                            // 384*384*2    =   294,912
#define WS_W2T  5480448                            // 384*384*2    =   294,912
#define WS_WFT  5775360                            // 128*384*2    =    98,304

extern "C" void kernel_launch(void* const* d_in, const int* in_sizes, int n_in,
                              void* d_out, int out_size, void* d_ws, size_t ws_size,
                              hipStream_t stream) {
  (void)in_sizes; (void)n_in; (void)out_size; (void)ws_size;
  const float*     node_embed = (const float*)d_in[0];
  const float*     edge_embed = (const float*)d_in[1];
  const long long* eidx       = (const long long*)d_in[2];   // int64 (2, N_EDGES)
  const float* W0 = (const float*)d_in[3];
  const float* b0 = (const float*)d_in[4];
  const float* W1 = (const float*)d_in[5];
  const float* b1 = (const float*)d_in[6];
  const float* W2 = (const float*)d_in[7];
  const float* b2 = (const float*)d_in[8];
  const float* Wf = (const float*)d_in[9];
  const float* bf = (const float*)d_in[10];
  const float* gamma = (const float*)d_in[11];
  const float* beta  = (const float*)d_in[12];
  float* out = (float*)d_out;

  char* ws = (char*)d_ws;
  unsigned short* hbf = (unsigned short*)(ws + WS_H);
  unsigned short* w0t = (unsigned short*)(ws + WS_W0T);
  unsigned short* w1t = (unsigned short*)(ws + WS_W1T);
  unsigned short* w2t = (unsigned short*)(ws + WS_W2T);
  unsigned short* wft = (unsigned short*)(ws + WS_WFT);

  cvt_transpose_kernel<<<(256 * 128 + 255) / 256, 256, 0, stream>>>(W0, w0t, 256, 128);
  cvt_transpose_kernel<<<(384 * 384 + 255) / 256, 256, 0, stream>>>(W1, w1t, 384, 384);
  cvt_transpose_kernel<<<(384 * 384 + 255) / 256, 256, 0, stream>>>(W2, w2t, 384, 384);
  cvt_transpose_kernel<<<(384 * 128 + 255) / 256, 256, 0, stream>>>(Wf, wft, 384, 128);

  node_proj_kernel<<<(N_NODES + MTILE - 1) / MTILE, THREADS, 0, stream>>>(node_embed, w0t, b0, hbf);

  edge_trunk_kernel<<<N_EDGES / MTILE, THREADS, SM_BYTES, stream>>>(
      edge_embed, eidx, hbf, w1t, b1, w2t, b2, wft, bf, gamma, beta, out);
}
